// EPLL_regularizer_66614942761138
// MI455X (gfx1250) — matrix operand order repaired and checked by reading, not compiled, not run on hardware
//
#include <hip/hip_runtime.h>
#include <math.h>

typedef __attribute__((ext_vector_type(16))) __bf16 v16bf;
typedef __attribute__((ext_vector_type(8)))  float v8f;

#define N_PTS 32768
#define KCOMP 200
#define DIM   64
#define NTILE 128
#define PSTR  36   // LDS row stride in u32 pairs: multiple of 4 (16B alignment), 36d%64 distinct for 16 rows

// ---- bf16 split helpers (round-to-nearest-even) ----------------------------
__device__ __forceinline__ unsigned f2bf(float x) {
  union { float f; unsigned u; } v; v.f = x;
  unsigned r = v.u + 0x7FFFu + ((v.u >> 16) & 1u);
  return r >> 16;                      // low 16 bits valid
}
__device__ __forceinline__ float bf2f(unsigned b) {
  union { unsigned u; float f; } v; v.u = b << 16;
  return v.f;
}
__device__ __forceinline__ void split2(float a, float b, unsigned& hi, unsigned& lo) {
  const unsigned ha = f2bf(a), hb = f2bf(b);
  const unsigned la = f2bf(a - bf2f(ha)), lb = f2bf(b - bf2f(hb));
  hi = ha | (hb << 16);
  lo = la | (lb << 16);
}
// Build a v16bf fragment from two 16B-aligned LDS chunks of 4 u32 each.
__device__ __forceinline__ v16bf load_frag(const unsigned* p0, const unsigned* p1) {
  union { int4 q[2]; v16bf v; } u;
  u.q[0] = *(const int4*)p0;
  u.q[1] = *(const int4*)p1;
  return u.v;
}

// ---------------------------------------------------------------------------
// Kernel 1: per-component Cholesky, triangular inverse, whitened mean, const.
// Emits Linv as packed bf16 hi/lo planes (pair-packed along e, row-major d).
// ---------------------------------------------------------------------------
__global__ __launch_bounds__(64) void gmm_prep(const float* __restrict__ covs,
                                               const float* __restrict__ means,
                                               const float* __restrict__ weights,
                                               unsigned* __restrict__ Lhi,
                                               unsigned* __restrict__ Llo,
                                               float* __restrict__ wm,
                                               float* __restrict__ ck) {
  __shared__ float L[DIM][DIM + 1];
  __shared__ float Inv[DIM][DIM + 1];
  const int k = blockIdx.x;
  const int tid = threadIdx.x;

  const float* C = covs + (size_t)k * DIM * DIM;
  for (int c = 0; c < DIM; ++c) L[tid][c] = C[(size_t)tid * DIM + c];
  __syncthreads();

  for (int j = 0; j < DIM; ++j) {            // column-by-column Cholesky (lower)
    if (tid == j) {
      float s = L[j][j];
      for (int m2 = 0; m2 < j; ++m2) s -= L[j][m2] * L[j][m2];
      L[j][j] = sqrtf(s);
    }
    __syncthreads();
    if (tid > j) {
      float s = L[tid][j];
      for (int m2 = 0; m2 < j; ++m2) s -= L[tid][m2] * L[j][m2];
      L[tid][j] = s / L[j][j];
    }
    __syncthreads();
  }

  // Triangular inverse: thread owns column tid (forward substitution)
  for (int i = 0; i < DIM; ++i) Inv[i][tid] = 0.0f;
  for (int i = tid; i < DIM; ++i) {
    float s = (i == tid) ? 1.0f : 0.0f;
    for (int m2 = tid; m2 < i; ++m2) s -= L[i][m2] * Inv[m2][tid];
    Inv[i][tid] = s / L[i][i];
  }
  __syncthreads();

  // Row tid of Linv -> bf16 hi/lo packed pairs; whitened mean (fp32)
  float acc = 0.0f;
  const float* mu = means + (size_t)k * DIM;
  unsigned* hrow = Lhi + (size_t)k * DIM * 32 + (size_t)tid * 32;
  unsigned* lrow = Llo + (size_t)k * DIM * 32 + (size_t)tid * 32;
  for (int p = 0; p < 32; ++p) {
    const float a = Inv[tid][2 * p];
    const float b = Inv[tid][2 * p + 1];
    unsigned hi, lo;
    split2(a, b, hi, lo);
    hrow[p] = hi;
    lrow[p] = lo;
    acc += a * mu[2 * p] + b * mu[2 * p + 1];
  }
  wm[k * DIM + tid] = acc;

  if (tid == 0) {
    float hl = 0.0f;
    for (int j = 0; j < DIM; ++j) hl += __logf(L[j][j]);
    float sw = 0.0f;
    for (int kk = 0; kk < KCOMP; ++kk) sw += weights[kk];
    const float LOG_2PI = 1.8378770664093453f;
    ck[k] = -0.5f * (float)DIM * LOG_2PI - hl + __logf(weights[k]) - __logf(sw);
  }
}

// ---------------------------------------------------------------------------
// Kernel 2: main compute, bf16x3 WMMA, double-buffered Linv staging.
// Block = 256 threads (8 wave32), 128 rows of X.
// The X staging LDS (2x128x36 u32) is reused as 2 double-buffered (hi+lo)
// Linv buffers (2 x 2 x 64 x 36 u32) after A-fragments move to registers.
// ---------------------------------------------------------------------------
__global__ __launch_bounds__(256) void gmm_main(const float* __restrict__ X,
                                                const unsigned* __restrict__ Lhi,
                                                const unsigned* __restrict__ Llo,
                                                const float* __restrict__ wm,
                                                const float* __restrict__ ck,
                                                float* __restrict__ ll) {
  __shared__ __align__(16) unsigned smem[2 * NTILE * PSTR];   // 9216 u32 = 36 KB
  __shared__ float swm2[2][DIM];

  const int tid  = threadIdx.x;
  const int wave = tid >> 5;
  const int lane = tid & 31;
  const int l16  = lane & 15;
  const int hlf  = lane >> 4;
  const int rowBase = blockIdx.x * NTILE;

  // --- Phase 1: X tile -> packed bf16 hi/lo pairs in LDS -------------------
  unsigned* sXh = smem;                   // [NTILE][PSTR]
  unsigned* sXl = smem + NTILE * PSTR;    // [NTILE][PSTR]
  for (int i = tid; i < NTILE * (DIM / 4); i += 256) {
    const int r = i >> 4;
    const int q = i & 15;                 // float4 index -> pairs 2q, 2q+1
    const float4 v = *(const float4*)(X + (size_t)(rowBase + r) * DIM + (q << 2));
    unsigned h0, l0, h1, l1;
    split2(v.x, v.y, h0, l0);
    split2(v.z, v.w, h1, l1);
    sXh[r * PSTR + 2 * q] = h0; sXh[r * PSTR + 2 * q + 1] = h1;
    sXl[r * PSTR + 2 * q] = l0; sXl[r * PSTR + 2 * q + 1] = l1;
  }
  __syncthreads();

  // A fragments (loop-invariant): 16x32 A in contiguous k = 16h + s order,
// per lane: VGPR0-7 = pairs 8h+0..7 (chunk base 16c)
  v16bf ahi[2], alo[2];
  {
    const unsigned* xrh = sXh + (wave * 16 + l16) * PSTR;
    const unsigned* xrl = sXl + (wave * 16 + l16) * PSTR;
#pragma unroll
    for (int c = 0; c < 2; ++c) {
      const int b0 = 16 * c + 8 * hlf;
      ahi[c] = load_frag(xrh + b0, xrh + b0 + 4);
      alo[c] = load_frag(xrl + b0, xrl + b0 + 4);
    }
  }
  __syncthreads();   // all waves done reading X staging before reuse

  // --- Phase 2: k-loop with double-buffered Linv planes --------------------
  // Each thread stages 2 uint4 per plane.
  const int i0 = tid,        r0 = i0 >> 3, p0 = (i0 & 7) << 2;
  const int i1 = tid + 256,  r1 = i1 >> 3, p1 = (i1 & 7) << 2;

  // Fill buffer 0 for k = 0
  {
    unsigned* b0h = smem;                  // buffer 0: hi plane
    unsigned* b0l = smem + DIM * PSTR;     // buffer 0: lo plane
    *(uint4*)(b0h + r0 * PSTR + p0) = *(const uint4*)(Lhi + (size_t)r0 * 32 + p0);
    *(uint4*)(b0h + r1 * PSTR + p1) = *(const uint4*)(Lhi + (size_t)r1 * 32 + p1);
    *(uint4*)(b0l + r0 * PSTR + p0) = *(const uint4*)(Llo + (size_t)r0 * 32 + p0);
    *(uint4*)(b0l + r1 * PSTR + p1) = *(const uint4*)(Llo + (size_t)r1 * 32 + p1);
    if (tid < DIM) swm2[0][tid] = wm[tid];
  }
  __syncthreads();

  float lse_m[8], lse_s[8];
#pragma unroll
  for (int r = 0; r < 8; ++r) { lse_m[r] = -INFINITY; lse_s[r] = 0.0f; }

  for (int k = 0; k < KCOMP; ++k) {
    const unsigned* curh = smem + (k & 1) * (2 * DIM * PSTR);
    const unsigned* curl = curh + DIM * PSTR;
    unsigned* nxth = smem + ((k + 1) & 1) * (2 * DIM * PSTR);
    unsigned* nxtl = nxth + DIM * PSTR;

    // Issue k+1 global loads FIRST so their latency overlaps the WMMA block.
    uint4 sh0 = {0,0,0,0}, sh1 = {0,0,0,0}, sl0 = {0,0,0,0}, sl1 = {0,0,0,0};
    float wnext = 0.0f;
    if (k + 1 < KCOMP) {
      const unsigned* LhiN = Lhi + (size_t)(k + 1) * DIM * 32;
      const unsigned* LloN = Llo + (size_t)(k + 1) * DIM * 32;
      sh0 = *(const uint4*)(LhiN + (size_t)r0 * 32 + p0);
      sh1 = *(const uint4*)(LhiN + (size_t)r1 * 32 + p1);
      sl0 = *(const uint4*)(LloN + (size_t)r0 * 32 + p0);
      sl1 = *(const uint4*)(LloN + (size_t)r1 * 32 + p1);
      if (tid < DIM) wnext = wm[(k + 1) * DIM + tid];
    }

    v8f acc[4];
#pragma unroll
    for (int j = 0; j < 4; ++j)
      acc[j] = (v8f){0.f, 0.f, 0.f, 0.f, 0.f, 0.f, 0.f, 0.f};

    // B 32x16 layout: lane l16 = N(=d), VGPR v / lane-half h -> K = 16h + 2v
    // -> pair index 16c + 8h + v in row d of packed Linv
#pragma unroll
    for (int c = 0; c < 2; ++c) {
#pragma unroll
      for (int j = 0; j < 4; ++j) {
        const unsigned* bh = curh + (j * 16 + l16) * PSTR + 16 * c + 8 * hlf;
        const unsigned* bl = curl + (j * 16 + l16) * PSTR + 16 * c + 8 * hlf;
        const v16bf bhi = load_frag(bh, bh + 4);
        const v16bf blo = load_frag(bl, bl + 4);
        acc[j] = __builtin_amdgcn_wmma_f32_16x16x32_bf16(
            false, ahi[c], false, bhi, (short)0, acc[j], false, false);
        acc[j] = __builtin_amdgcn_wmma_f32_16x16x32_bf16(
            false, alo[c], false, bhi, (short)0, acc[j], false, false);
        acc[j] = __builtin_amdgcn_wmma_f32_16x16x32_bf16(
            false, ahi[c], false, blo, (short)0, acc[j], false, false);
      }
    }

    // q partials: acc[j][r] = W[row = wave*16 + r + 8*hlf][d = 16j + l16]
    float qp[8];
#pragma unroll
    for (int r = 0; r < 8; ++r) qp[r] = 0.0f;
#pragma unroll
    for (int j = 0; j < 4; ++j) {
      const float wv = swm2[k & 1][j * 16 + l16];
#pragma unroll
      for (int r = 0; r < 8; ++r) {
        const float d = acc[j][r] - wv;
        qp[r] = fmaf(d, d, qp[r]);
      }
    }
#pragma unroll
    for (int off = 1; off < 16; off <<= 1) {
#pragma unroll
      for (int r = 0; r < 8; ++r) qp[r] += __shfl_xor(qp[r], off, 32);
    }

    // Streaming logsumexp
    const float c0 = ck[k];
#pragma unroll
    for (int r = 0; r < 8; ++r) {
      const float lp = fmaf(-0.5f, qp[r], c0);
      const float nm = fmaxf(lse_m[r], lp);
      lse_s[r] = lse_s[r] * __expf(lse_m[r] - nm) + __expf(lp - nm);
      lse_m[r] = nm;
    }

    // Drain staged k+1 data into the alternate buffer (disjoint from curh/curl)
    if (k + 1 < KCOMP) {
      *(uint4*)(nxth + r0 * PSTR + p0) = sh0;
      *(uint4*)(nxth + r1 * PSTR + p1) = sh1;
      *(uint4*)(nxtl + r0 * PSTR + p0) = sl0;
      *(uint4*)(nxtl + r1 * PSTR + p1) = sl1;
      if (tid < DIM) swm2[(k + 1) & 1][tid] = wnext;
    }
    __syncthreads();   // single barrier per iteration
  }

  if (l16 == 0) {
#pragma unroll
    for (int r = 0; r < 8; ++r)
      ll[rowBase + wave * 16 + 8 * hlf + r] = lse_m[r] + __logf(lse_s[r]);
  }
}

// ---------------------------------------------------------------------------
// Kernel 3: final -mean reduction (single block)
// ---------------------------------------------------------------------------
__global__ __launch_bounds__(256) void gmm_reduce(const float* __restrict__ ll,
                                                  float* __restrict__ out) {
  __shared__ float red[256];
  float s = 0.0f;
  for (int i = threadIdx.x; i < N_PTS; i += 256) s += ll[i];
  red[threadIdx.x] = s;
  __syncthreads();
  for (int st = 128; st > 0; st >>= 1) {
    if (threadIdx.x < st) red[threadIdx.x] += red[threadIdx.x + st];
    __syncthreads();
  }
  if (threadIdx.x == 0) out[0] = -red[0] / (float)N_PTS;
}

extern "C" void kernel_launch(void* const* d_in, const int* in_sizes, int n_in,
                              void* d_out, int out_size, void* d_ws, size_t ws_size,
                              hipStream_t stream) {
  const float* fake_data = (const float*)d_in[0];  // (N, D)
  const float* weights   = (const float*)d_in[1];  // (K,)
  const float* means     = (const float*)d_in[2];  // (K, D)
  const float* covs      = (const float*)d_in[3];  // (K, D, D)

  unsigned* Lhi = (unsigned*)d_ws;                          // K*64*32 u32
  unsigned* Llo = Lhi + (size_t)KCOMP * DIM * 32;           // K*64*32 u32
  float* wmv = (float*)(Llo + (size_t)KCOMP * DIM * 32);    // K*D
  float* ckv = wmv + (size_t)KCOMP * DIM;                   // K (pad to 256)
  float* ll  = ckv + 256;                                   // N

  gmm_prep<<<KCOMP, 64, 0, stream>>>(covs, means, weights, Lhi, Llo, wmv, ckv);
  gmm_main<<<N_PTS / NTILE, 256, 0, stream>>>(fake_data, Lhi, Llo, wmv, ckv, ll);
  gmm_reduce<<<1, 256, 0, stream>>>(ll, (float*)d_out);
}
